// OpcodeMoELayer_84000970375604
// MI455X (gfx1250) — compile-verified, matrix operand order checked by reading, not run on hardware
//
#include <hip/hip_runtime.h>
#include <hip/hip_bf16.h>

// ---------- problem dims ----------
constexpr int Bsz = 16, Sdim = 2048, Ddim = 1024, Edim = 8, Hdim = 2048;

// ---------- tiling ----------
constexpr int TM = 128;   // block tile M
constexpr int TN = 128;   // block tile N
constexpr int TK = 32;    // K step (bf16 WMMA K)
constexpr int LDS_STRIDE = TK + 8;  // 40 bf16 = 80B = 20 banks -> conflict-free b128 reads

// ---------- types ----------
typedef __attribute__((ext_vector_type(16))) __bf16        v16bf;
typedef __attribute__((ext_vector_type(8)))  float         v8f;
typedef __attribute__((ext_vector_type(4)))  unsigned int  u32x4;
typedef __attribute__((ext_vector_type(4)))  int           i32x4;
typedef __attribute__((ext_vector_type(4)))  float         f32x4;
typedef __attribute__((ext_vector_type(4)))  unsigned short u16x4;

union FragBF {       // 16 bf16 = one WMMA A/B operand, loaded as 2x16B chunks
    u32x4 u[2];
    v16bf v;
};

// ---------- CDNA5 async global->LDS path (guarded; fallback = sync staging) ----------
#if defined(__AMDGCN__) && \
    __has_builtin(__builtin_amdgcn_global_load_async_to_lds_b128) && \
    __has_builtin(__builtin_amdgcn_s_wait_asynccnt)
#define USE_ASYNC 1
typedef __attribute__((address_space(1))) i32x4* gi32x4_p;   // global v4i32*
typedef __attribute__((address_space(3))) i32x4* li32x4_p;   // LDS    v4i32*
#define ASYNC_CP16(gp, lp)                                                   \
    __builtin_amdgcn_global_load_async_to_lds_b128(                          \
        (gi32x4_p)(gp), (li32x4_p)(lp), 0, 0)
#else
#define USE_ASYNC 0
#endif

static __device__ __forceinline__ unsigned short f2bf(float f) {
    unsigned int u = __float_as_uint(f);
    u += 0x7FFFu + ((u >> 16) & 1u);   // round-to-nearest-even
    return (unsigned short)(u >> 16);
}

static __device__ __forceinline__ float silu(float x) {
    return x / (1.0f + __expf(-x));
}

// ---------------------------------------------------------------------------
// expert selection: argmax over opcode_onehot [B,E]
// ---------------------------------------------------------------------------
__global__ void expert_select_kernel(const float* __restrict__ oh, int* __restrict__ experts) {
    int b = threadIdx.x;
    if (b < Bsz) {
        float best = oh[b * Edim];
        int bi = 0;
        #pragma unroll
        for (int e = 1; e < Edim; ++e) {
            float v = oh[b * Edim + e];
            if (v > best) { best = v; bi = e; }
        }
        experts[b] = bi;
    }
}

// ---------------------------------------------------------------------------
// fp32 -> bf16 conversion, 4 elements per thread iteration
// ---------------------------------------------------------------------------
__global__ __launch_bounds__(256)
void cvt_f32_bf16_kernel(const float* __restrict__ src, unsigned short* __restrict__ dst, int n4) {
    for (int i = blockIdx.x * blockDim.x + threadIdx.x; i < n4; i += gridDim.x * blockDim.x) {
        f32x4 f = ((const f32x4*)src)[i];
        u16x4 o;
        o.x = f2bf(f.x); o.y = f2bf(f.y); o.z = f2bf(f.z); o.w = f2bf(f.w);
        ((u16x4*)dst)[i] = o;
    }
}

// ---------------------------------------------------------------------------
// WMMA micro-steps (fragment fetch per ISA 16-bit A layout: lane = m + 16*kh,
// two 16B chunks at K offsets 0 and +16 elements)
// ---------------------------------------------------------------------------
static __device__ __forceinline__ void wmma_step_ug(
    const unsigned short* __restrict__ lA, const unsigned short* __restrict__ lU,
    const unsigned short* __restrict__ lG,
    int wm, int wn, int lm, int kh, v8f accU[4][2], v8f accG[4][2])
{
    FragBF fa[4], fu[2], fg[2];
    #pragma unroll
    for (int i = 0; i < 4; ++i) {
        const unsigned short* p = lA + (wm * 64 + i * 16 + lm) * LDS_STRIDE + kh * 8;
        fa[i].u[0] = *(const u32x4*)p;
        fa[i].u[1] = *(const u32x4*)(p + 16);
    }
    #pragma unroll
    for (int j = 0; j < 2; ++j) {
        const unsigned short* pu = lU + (wn * 32 + j * 16 + lm) * LDS_STRIDE + kh * 8;
        const unsigned short* pg = lG + (wn * 32 + j * 16 + lm) * LDS_STRIDE + kh * 8;
        fu[j].u[0] = *(const u32x4*)pu;  fu[j].u[1] = *(const u32x4*)(pu + 16);
        fg[j].u[0] = *(const u32x4*)pg;  fg[j].u[1] = *(const u32x4*)(pg + 16);
    }
    #pragma unroll
    for (int i = 0; i < 4; ++i)
        #pragma unroll
        for (int j = 0; j < 2; ++j) {
            accU[i][j] = __builtin_amdgcn_wmma_f32_16x16x32_bf16(
                false, fa[i].v, false, fu[j].v, (short)0, accU[i][j], false, false);
            accG[i][j] = __builtin_amdgcn_wmma_f32_16x16x32_bf16(
                false, fa[i].v, false, fg[j].v, (short)0, accG[i][j], false, false);
        }
}

static __device__ __forceinline__ void wmma_step_dn(
    const unsigned short* __restrict__ lA, const unsigned short* __restrict__ lB,
    int wm, int wn, int lm, int kh, v8f acc[4][2])
{
    FragBF fa[4], fb[2];
    #pragma unroll
    for (int i = 0; i < 4; ++i) {
        const unsigned short* p = lA + (wm * 64 + i * 16 + lm) * LDS_STRIDE + kh * 8;
        fa[i].u[0] = *(const u32x4*)p;
        fa[i].u[1] = *(const u32x4*)(p + 16);
    }
    #pragma unroll
    for (int j = 0; j < 2; ++j) {
        const unsigned short* p = lB + (wn * 32 + j * 16 + lm) * LDS_STRIDE + kh * 8;
        fb[j].u[0] = *(const u32x4*)p;
        fb[j].u[1] = *(const u32x4*)(p + 16);
    }
    #pragma unroll
    for (int i = 0; i < 4; ++i)
        #pragma unroll
        for (int j = 0; j < 2; ++j)
            acc[i][j] = __builtin_amdgcn_wmma_f32_16x16x32_bf16(
                false, fa[i].v, false, fb[j].v, (short)0, acc[i][j], false, false);
}

// ---------------------------------------------------------------------------
// Fused up+gate GEMM + SwiGLU, double-buffered LDS, async global->LDS staging.
//   A = x_bf16[b] [S,D], Bu = W_up_bf16[e] [H,D], Bg = W_gate_bf16[e] [H,D]
//   hidden[b,s,h] = silu(A.Bu^T + b_up) * (A.Bg^T + b_gate)   (stored bf16)
// 256 threads = 8 waves (2M x 4N); wave tile 64x32 = 4x2 WMMA subtiles.
// ---------------------------------------------------------------------------
__global__ __launch_bounds__(256)
void moe_upgate_kernel(const unsigned short* __restrict__ xb,
                       const unsigned short* __restrict__ wub,
                       const unsigned short* __restrict__ wgb,
                       const float* __restrict__ b_up,
                       const float* __restrict__ b_gate,
                       const int* __restrict__ experts,
                       unsigned short* __restrict__ hidden)
{
    __shared__ unsigned short lA[2][TM * LDS_STRIDE];
    __shared__ unsigned short lU[2][TN * LDS_STRIDE];
    __shared__ unsigned short lG[2][TN * LDS_STRIDE];

    const int n0 = blockIdx.x * TN;
    const int s0 = blockIdx.y * TM;
    const int b  = blockIdx.z;
    const int e  = experts[b];

    const int tid  = threadIdx.x;
    const int lane = tid & 31;
    const int wave = tid >> 5;
    const int wm   = wave & 1;
    const int wn   = wave >> 1;
    const int lm   = lane & 15;
    const int kh   = lane >> 4;

    // staging assignment: thread covers rows r0 and r0+64, 8-elem chunk sb
    const int r0 = tid >> 2;
    const int sb = tid & 3;

    const unsigned short* xrow = xb  + (size_t)b * Sdim * Ddim;
    const unsigned short* urow = wub + (size_t)e * Hdim * Ddim;
    const unsigned short* grow = wgb + (size_t)e * Hdim * Ddim;

    v8f accU[4][2], accG[4][2];
    #pragma unroll
    for (int i = 0; i < 4; ++i)
        #pragma unroll
        for (int j = 0; j < 2; ++j) { accU[i][j] = (v8f)0.0f; accG[i][j] = (v8f)0.0f; }

    constexpr int NK = Ddim / TK;

#if USE_ASYNC
    #define STAGE_UG(kk, buf)                                                             \
        do {                                                                              \
            ASYNC_CP16(xrow + (size_t)(s0 + r0)      * Ddim + (kk) + sb * 8,              \
                       lA[buf] + r0        * LDS_STRIDE + sb * 8);                        \
            ASYNC_CP16(xrow + (size_t)(s0 + r0 + 64) * Ddim + (kk) + sb * 8,              \
                       lA[buf] + (r0 + 64) * LDS_STRIDE + sb * 8);                        \
            ASYNC_CP16(urow + (size_t)(n0 + r0)      * Ddim + (kk) + sb * 8,              \
                       lU[buf] + r0        * LDS_STRIDE + sb * 8);                        \
            ASYNC_CP16(urow + (size_t)(n0 + r0 + 64) * Ddim + (kk) + sb * 8,              \
                       lU[buf] + (r0 + 64) * LDS_STRIDE + sb * 8);                        \
            ASYNC_CP16(grow + (size_t)(n0 + r0)      * Ddim + (kk) + sb * 8,              \
                       lG[buf] + r0        * LDS_STRIDE + sb * 8);                        \
            ASYNC_CP16(grow + (size_t)(n0 + r0 + 64) * Ddim + (kk) + sb * 8,              \
                       lG[buf] + (r0 + 64) * LDS_STRIDE + sb * 8);                        \
        } while (0)

    STAGE_UG(0, 0);
    for (int it = 0; it < NK; ++it) {
        const int cur = it & 1;
        if (it + 1 < NK) {
            STAGE_UG((it + 1) * TK, cur ^ 1);
            __builtin_amdgcn_s_wait_asynccnt(6);  // in-order: drains the 6 for `cur`
        } else {
            __builtin_amdgcn_s_wait_asynccnt(0);
        }
        __syncthreads();                           // cur buffer visible to all waves
        wmma_step_ug(lA[cur], lU[cur], lG[cur], wm, wn, lm, kh, accU, accG);
        __syncthreads();                           // done reading before next overwrite
    }
    #undef STAGE_UG
#else
    for (int it = 0; it < NK; ++it) {
        const int kk = it * TK;
        u32x4 ra0, ra1, ru0, ru1, rg0, rg1;
        ra0 = *(const u32x4*)(xrow + (size_t)(s0 + r0)      * Ddim + kk + sb * 8);
        ra1 = *(const u32x4*)(xrow + (size_t)(s0 + r0 + 64) * Ddim + kk + sb * 8);
        ru0 = *(const u32x4*)(urow + (size_t)(n0 + r0)      * Ddim + kk + sb * 8);
        ru1 = *(const u32x4*)(urow + (size_t)(n0 + r0 + 64) * Ddim + kk + sb * 8);
        rg0 = *(const u32x4*)(grow + (size_t)(n0 + r0)      * Ddim + kk + sb * 8);
        rg1 = *(const u32x4*)(grow + (size_t)(n0 + r0 + 64) * Ddim + kk + sb * 8);
        __syncthreads();
        *(u32x4*)(lA[0] + r0        * LDS_STRIDE + sb * 8) = ra0;
        *(u32x4*)(lA[0] + (r0 + 64) * LDS_STRIDE + sb * 8) = ra1;
        *(u32x4*)(lU[0] + r0        * LDS_STRIDE + sb * 8) = ru0;
        *(u32x4*)(lU[0] + (r0 + 64) * LDS_STRIDE + sb * 8) = ru1;
        *(u32x4*)(lG[0] + r0        * LDS_STRIDE + sb * 8) = rg0;
        *(u32x4*)(lG[0] + (r0 + 64) * LDS_STRIDE + sb * 8) = rg1;
        __syncthreads();
        wmma_step_ug(lA[0], lU[0], lG[0], wm, wn, lm, kh, accU, accG);
    }
#endif

    // ---- epilogue: SwiGLU, bf16 store ----
    unsigned short* hrow = hidden + (size_t)b * Sdim * Hdim;
    #pragma unroll
    for (int i = 0; i < 4; ++i)
        #pragma unroll
        for (int j = 0; j < 2; ++j)
            #pragma unroll
            for (int r = 0; r < 8; ++r) {
                int m = wm * 64 + i * 16 + r + kh * 8;   // C layout: lanes16-31 -> M+8
                int n = wn * 32 + j * 16 + lm;
                float up = accU[i][j][r] + b_up[e * Hdim + n0 + n];
                float gt = accG[i][j][r] + b_gate[e * Hdim + n0 + n];
                hrow[(size_t)(s0 + m) * Hdim + (n0 + n)] = f2bf(silu(up) * gt);
            }
}

// ---------------------------------------------------------------------------
// Down GEMM: out[b,s,d] = hidden[b,s,:] . W_down[e][d,:] + b_down[e][d]
//   A = hidden[b] [S,H], B = W_down_bf16[e] [D,H], K = H
// ---------------------------------------------------------------------------
__global__ __launch_bounds__(256)
void moe_down_kernel(const unsigned short* __restrict__ hidden,
                     const unsigned short* __restrict__ wdb,
                     const float* __restrict__ b_down,
                     const int* __restrict__ experts,
                     float* __restrict__ out)
{
    __shared__ unsigned short lA[2][TM * LDS_STRIDE];
    __shared__ unsigned short lB[2][TN * LDS_STRIDE];

    const int n0 = blockIdx.x * TN;   // over D
    const int s0 = blockIdx.y * TM;   // over S
    const int b  = blockIdx.z;
    const int e  = experts[b];

    const int tid  = threadIdx.x;
    const int lane = tid & 31;
    const int wave = tid >> 5;
    const int wm   = wave & 1;
    const int wn   = wave >> 1;
    const int lm   = lane & 15;
    const int kh   = lane >> 4;

    const int r0 = tid >> 2;
    const int sb = tid & 3;

    const unsigned short* arow = hidden + (size_t)b * Sdim * Hdim;
    const unsigned short* brow = wdb    + (size_t)e * Ddim * Hdim;

    v8f acc[4][2];
    #pragma unroll
    for (int i = 0; i < 4; ++i)
        #pragma unroll
        for (int j = 0; j < 2; ++j) acc[i][j] = (v8f)0.0f;

    constexpr int NK = Hdim / TK;

#if USE_ASYNC
    #define STAGE_DN(kk, buf)                                                             \
        do {                                                                              \
            ASYNC_CP16(arow + (size_t)(s0 + r0)      * Hdim + (kk) + sb * 8,              \
                       lA[buf] + r0        * LDS_STRIDE + sb * 8);                        \
            ASYNC_CP16(arow + (size_t)(s0 + r0 + 64) * Hdim + (kk) + sb * 8,              \
                       lA[buf] + (r0 + 64) * LDS_STRIDE + sb * 8);                        \
            ASYNC_CP16(brow + (size_t)(n0 + r0)      * Hdim + (kk) + sb * 8,              \
                       lB[buf] + r0        * LDS_STRIDE + sb * 8);                        \
            ASYNC_CP16(brow + (size_t)(n0 + r0 + 64) * Hdim + (kk) + sb * 8,              \
                       lB[buf] + (r0 + 64) * LDS_STRIDE + sb * 8);                        \
        } while (0)

    STAGE_DN(0, 0);
    for (int it = 0; it < NK; ++it) {
        const int cur = it & 1;
        if (it + 1 < NK) {
            STAGE_DN((it + 1) * TK, cur ^ 1);
            __builtin_amdgcn_s_wait_asynccnt(4);
        } else {
            __builtin_amdgcn_s_wait_asynccnt(0);
        }
        __syncthreads();
        wmma_step_dn(lA[cur], lB[cur], wm, wn, lm, kh, acc);
        __syncthreads();
    }
    #undef STAGE_DN
#else
    for (int it = 0; it < NK; ++it) {
        const int kk = it * TK;
        u32x4 ra0, ra1, rb0, rb1;
        ra0 = *(const u32x4*)(arow + (size_t)(s0 + r0)      * Hdim + kk + sb * 8);
        ra1 = *(const u32x4*)(arow + (size_t)(s0 + r0 + 64) * Hdim + kk + sb * 8);
        rb0 = *(const u32x4*)(brow + (size_t)(n0 + r0)      * Hdim + kk + sb * 8);
        rb1 = *(const u32x4*)(brow + (size_t)(n0 + r0 + 64) * Hdim + kk + sb * 8);
        __syncthreads();
        *(u32x4*)(lA[0] + r0        * LDS_STRIDE + sb * 8) = ra0;
        *(u32x4*)(lA[0] + (r0 + 64) * LDS_STRIDE + sb * 8) = ra1;
        *(u32x4*)(lB[0] + r0        * LDS_STRIDE + sb * 8) = rb0;
        *(u32x4*)(lB[0] + (r0 + 64) * LDS_STRIDE + sb * 8) = rb1;
        __syncthreads();
        wmma_step_dn(lA[0], lB[0], wm, wn, lm, kh, acc);
    }
#endif

    float* orow = out + (size_t)b * Sdim * Ddim;
    #pragma unroll
    for (int i = 0; i < 4; ++i)
        #pragma unroll
        for (int j = 0; j < 2; ++j)
            #pragma unroll
            for (int r = 0; r < 8; ++r) {
                int m = wm * 64 + i * 16 + r + kh * 8;
                int n = wn * 32 + j * 16 + lm;
                orow[(size_t)(s0 + m) * Ddim + (n0 + n)] =
                    acc[i][j][r] + b_down[e * Ddim + n0 + n];
            }
}

// ---------------------------------------------------------------------------
extern "C" void kernel_launch(void* const* d_in, const int* in_sizes, int n_in,
                              void* d_out, int out_size, void* d_ws, size_t ws_size,
                              hipStream_t stream) {
    const float* x      = (const float*)d_in[0];
    const float* onehot = (const float*)d_in[1];
    const float* W_up   = (const float*)d_in[2];
    const float* b_up   = (const float*)d_in[3];
    const float* W_gate = (const float*)d_in[4];
    const float* b_gate = (const float*)d_in[5];
    const float* W_down = (const float*)d_in[6];
    const float* b_down = (const float*)d_in[7];
    float* out = (float*)d_out;

    // workspace layout
    char* ws = (char*)d_ws;
    int* experts = (int*)ws;                                    // 16 ints
    const size_t xN = (size_t)Bsz * Sdim * Ddim;                // 33.5M elems
    const size_t wN = (size_t)Edim * Hdim * Ddim;               // 16.8M elems
    unsigned short* xb  = (unsigned short*)(ws + 256);
    unsigned short* wub = xb  + xN;
    unsigned short* wgb = wub + wN;
    unsigned short* wdb = wgb + wN;
    unsigned short* hid = wdb + wN;
    (void)in_sizes; (void)n_in; (void)out_size; (void)ws_size;

    expert_select_kernel<<<1, 32, 0, stream>>>(onehot, experts);

    cvt_f32_bf16_kernel<<<2048, 256, 0, stream>>>(x,      xb,  (int)(xN / 4));
    cvt_f32_bf16_kernel<<<2048, 256, 0, stream>>>(W_up,   wub, (int)(wN / 4));
    cvt_f32_bf16_kernel<<<2048, 256, 0, stream>>>(W_gate, wgb, (int)(wN / 4));
    cvt_f32_bf16_kernel<<<2048, 256, 0, stream>>>(W_down, wdb, (int)(wN / 4));

    dim3 gA(Hdim / TN, Sdim / TM, Bsz);    // 16 x 16 x 16
    moe_upgate_kernel<<<gA, 256, 0, stream>>>(xb, wub, wgb, b_up, b_gate, experts, hid);

    dim3 gB(Ddim / TN, Sdim / TM, Bsz);    // 8 x 16 x 16
    moe_down_kernel<<<gB, 256, 0, stream>>>(hid, wdb, b_down, experts, out);
}
